// OmniTransformer_v5_43722767073430
// MI455X (gfx1250) — matrix operand order repaired and checked by reading, not compiled, not run on hardware
//
#include <hip/hip_runtime.h>
#include <hip/hip_bf16.h>
#include <math.h>

typedef __attribute__((ext_vector_type(16))) __bf16 v16bf;
typedef __attribute__((ext_vector_type(8)))  __bf16 v8bf;
typedef __attribute__((ext_vector_type(8)))  float  v8f;

// ---------------------------------------------------------------------------
// Fragment loaders (bf16 source, per ISA 7.12.2 layouts):
// A-frag (16x32): lane row = lane&15; element e -> K = (e/8)*16 + (lane>>4)*8 + e%8
//   => two contiguous 8-elem runs at +hi*8 and +16+hi*8.
// B-frag (32x16): lane col = lane&15; element e -> K = (e/8)*16 + (lane>>4)*8 + e%8 (same as A)
//   => two contiguous 8-elem runs at +hi*8 and +16+hi*8.
// ---------------------------------------------------------------------------
__device__ inline v16bf frag_bf16(const __bf16* __restrict__ p0,
                                  const __bf16* __restrict__ p1) {
  v8bf lo = *(const v8bf*)p0;
  v8bf hh = *(const v8bf*)p1;
  v16bf f;
#pragma unroll
  for (int e = 0; e < 8; ++e) { f[e] = lo[e]; f[e + 8] = hh[e]; }
  return f;
}

__device__ inline float gelu_exact(float v) {
  return 0.5f * v * (1.0f + erff(v * 0.70710678118f));
}

// ---------------------------------------------------------------------------
// f32 -> bf16 conversion (one-time weight pre-pass).
// ---------------------------------------------------------------------------
__global__ __launch_bounds__(256)
void cvt_f32_bf16(const float* __restrict__ in, __bf16* __restrict__ out, long n) {
  long i = (long)blockIdx.x * 256 + threadIdx.x;
  if (i < n) out[i] = (__bf16)in[i];
}

// ---------------------------------------------------------------------------
// GEMM: Out[M,N] = A[M,K]*Wt[N,K]^T (+bias)(+gelu)(+res)(+vec); A,Wt bf16.
// Block = 256 thr = 8 waves; block tile 256x32; wave tile 32x32
// (2 A-frags x 2 B-frags -> 4 WMMA per k-step, zero in-loop converts).
// flags: 1=gelu, 2=add res[M,N] (f32), 4=add vec[N] (f32), 8=store bf16
// ---------------------------------------------------------------------------
__global__ __launch_bounds__(256)
void gemm_bf16_wmma(const __bf16* __restrict__ A, const __bf16* __restrict__ Wt,
                    const float* __restrict__ bias, const float* __restrict__ res,
                    const float* __restrict__ vec, void* __restrict__ OutV,
                    int M, int N, int K, int flags) {
  const int lane = threadIdx.x & 31;
  const int wave = threadIdx.x >> 5;
  const int hi = lane >> 4;
  const int lcol = lane & 15;
  const int m0 = blockIdx.x * 256 + wave * 32;
  const int n0 = blockIdx.y * 32;

  int ar0 = m0 + lcol;      if (ar0 >= M) ar0 = M - 1;
  int ar1 = m0 + 16 + lcol; if (ar1 >= M) ar1 = M - 1;
  const __bf16* A0 = A  + (size_t)ar0 * K;
  const __bf16* A1 = A  + (size_t)ar1 * K;
  const __bf16* B0 = Wt + (size_t)(n0 + lcol) * K;
  const __bf16* B1 = Wt + (size_t)(n0 + 16 + lcol) * K;

  v8f c00{}; v8f c01{}; v8f c10{}; v8f c11{};
  for (int kb = 0; kb < K; kb += 32) {
    if (kb + 32 < K) {
      __builtin_prefetch(A0 + kb + 32, 0, 3);
      __builtin_prefetch(A1 + kb + 32, 0, 3);
      __builtin_prefetch(B0 + kb + 32, 0, 3);
      __builtin_prefetch(B1 + kb + 32, 0, 3);
    }
    v16bf a0 = frag_bf16(A0 + kb + hi * 8,  A0 + kb + 16 + hi * 8);
    v16bf a1 = frag_bf16(A1 + kb + hi * 8,  A1 + kb + 16 + hi * 8);
    v16bf b0 = frag_bf16(B0 + kb + hi * 8,  B0 + kb + 16 + hi * 8);
    v16bf b1 = frag_bf16(B1 + kb + hi * 8,  B1 + kb + 16 + hi * 8);
    c00 = __builtin_amdgcn_wmma_f32_16x16x32_bf16(false, a0, false, b0, (short)0, c00, false, false);
    c01 = __builtin_amdgcn_wmma_f32_16x16x32_bf16(false, a0, false, b1, (short)0, c01, false, false);
    c10 = __builtin_amdgcn_wmma_f32_16x16x32_bf16(false, a1, false, b0, (short)0, c10, false, false);
    c11 = __builtin_amdgcn_wmma_f32_16x16x32_bf16(false, a1, false, b1, (short)0, c11, false, false);
  }

  float* Outf = (float*)OutV;
  __bf16* Outh = (__bf16*)OutV;
#pragma unroll
  for (int q = 0; q < 4; ++q) {
    v8f acc = (q == 0) ? c00 : (q == 1) ? c01 : (q == 2) ? c10 : c11;
    int rbase = m0 + ((q >> 1) ? 16 : 0);
    int col = n0 + ((q & 1) ? 16 : 0) + lcol;
#pragma unroll
    for (int r8 = 0; r8 < 8; ++r8) {
      int row = rbase + hi * 8 + r8;
      if (row < M) {
        float v = acc[r8] + (bias ? bias[col] : 0.0f);
        if (flags & 1) v = gelu_exact(v);
        if (flags & 2) v += res[(size_t)row * N + col];
        if (flags & 4) v += vec[col];
        if (flags & 8) Outh[(size_t)row * N + col] = (__bf16)v;
        else           Outf[(size_t)row * N + col] = v;
      }
    }
  }
}

// ---------------------------------------------------------------------------
// Region label for shifted-window mask.
// ---------------------------------------------------------------------------
__device__ inline int lbl1(int g, int s, int wsz, int z) {
  if (z == 0) return 0;
  if (g < s - wsz) return 0;
  if (g < s - z) return 1;
  return 2;
}

// ---------------------------------------------------------------------------
// Windowed self-attention: WMMA QK^T + flash online softmax + WMMA PV.
// One WG per (window, head). N=392, hd=32. qkv is bf16, out is bf16.
// 1/sqrt(hd) folded into f32 logits post-WMMA.
// ---------------------------------------------------------------------------
__global__ __launch_bounds__(256)
void win_attn(const __bf16* __restrict__ qkv, __bf16* __restrict__ out,
              const float* __restrict__ rel, int heads, int C,
              int T, int H, int W, int s0, int s1, int s2) {
  const int head = blockIdx.x % heads;
  const int win  = blockIdx.x / heads;
  const int lane = threadIdx.x & 31;
  const int wave = threadIdx.x >> 5;
  const int hi   = lane >> 4;
  const int lcol = lane & 15;
  const int C3   = 3 * C;
  const int nWw = W / 7, nWh = H / 7;
  const int ww_ = win % nWw, hw_ = (win / nWw) % nWh, tw_ = win / (nWw * nWh);
  const int shifted = (s0 | s1 | s2);

  __shared__ __align__(16) __bf16 Vt[32 * 416];      // V^T [hd][tok], padded 416
  __shared__ __align__(16) __bf16 Pld[8 * 16 * 32];  // per-wave P transpose staging

  const size_t wbase = (size_t)win * 392 * C3;
  for (int i = threadIdx.x; i < 32 * 416; i += 256) {
    int d = i / 416, tok = i % 416;
    Vt[d * 416 + tok] = (tok < 392)
        ? qkv[wbase + (size_t)tok * C3 + 2 * C + head * 32 + d] : (__bf16)0.0f;
  }
  __syncthreads();

  const float scale = 0.1767766953f;  // 1/sqrt(32)

  for (int rt = wave; rt < 25; rt += 8) {
    const int m0 = rt * 16;

    int qrow = m0 + lcol; if (qrow > 391) qrow = 391;
    const __bf16* qb = qkv + wbase + (size_t)qrow * C3 + head * 32;
    v16bf qf = frag_bf16(qb + hi * 8, qb + 16 + hi * 8);

    int ti[8], hh_[8], wi_[8], li[8];
#pragma unroll
    for (int r8 = 0; r8 < 8; ++r8) {
      int i = m0 + hi * 8 + r8; if (i > 391) i = 391;
      int t = i / 49, hl = (i % 49) / 7, wl = i % 7;
      ti[r8] = t; hh_[r8] = hl; wi_[r8] = wl;
      int gt = tw_ * 8 + t, gh = hw_ * 7 + hl, gw = ww_ * 7 + wl;
      li[r8] = (lbl1(gt, T, 8, s0) * 3 + lbl1(gh, H, 7, s1)) * 3 + lbl1(gw, W, 7, s2);
    }

    float m_r[8], s_r[8]; v8f o0{}; v8f o1{};
#pragma unroll
    for (int r8 = 0; r8 < 8; ++r8) { m_r[r8] = -1e30f; s_r[r8] = 0.0f; }

    for (int cp = 0; cp < 13; ++cp) {  // 13 x 32 cols = 416 (>=392)
      const int j0 = cp * 32;
      float sv[2][8];

      for (int half = 0; half < 2; ++half) {
        const int jb = j0 + half * 16;
        int ktok = jb + lcol; if (ktok > 391) ktok = 391;
        const __bf16* kbp = qkv + wbase + (size_t)ktok * C3 + C + head * 32;
        v16bf kf = frag_bf16(kbp + hi * 8, kbp + 16 + hi * 8);
        v8f z{};
        v8f s = __builtin_amdgcn_wmma_f32_16x16x32_bf16(false, qf, false, kf, (short)0, z, false, false);

        const int j = jb + lcol;
        int jj = (j > 391) ? 391 : j;
        int tj = jj / 49, hj = (jj % 49) / 7, wj = jj % 7;
        int lj = 0;
        if (shifted) {
          int gt = tw_ * 8 + tj, gh = hw_ * 7 + hj, gw = ww_ * 7 + wj;
          lj = (lbl1(gt, T, 8, s0) * 3 + lbl1(gh, H, 7, s1)) * 3 + lbl1(gw, W, 7, s2);
        }
#pragma unroll
        for (int r8 = 0; r8 < 8; ++r8) {
          float v;
          if (j > 391) v = -1e30f;
          else {
            int idxr = (ti[r8] - tj + 7) * 169 + (hh_[r8] - hj + 6) * 13 + (wi_[r8] - wj + 6);
            v = s[r8] * scale + rel[(size_t)idxr * heads + head];
            if (shifted && (li[r8] != lj)) v -= 100.0f;
          }
          sv[half][r8] = v;
        }
      }

      // online softmax over the 32-col slab
#pragma unroll
      for (int r8 = 0; r8 < 8; ++r8) {
        float mx = fmaxf(sv[0][r8], sv[1][r8]);
        for (int off = 1; off < 16; off <<= 1) mx = fmaxf(mx, __shfl_xor(mx, off, 16));
        float mn = fmaxf(m_r[r8], mx);
        float sc = __expf(m_r[r8] - mn);
        float p0 = (sv[0][r8] > -1e29f) ? __expf(sv[0][r8] - mn) : 0.0f;
        float p1 = (sv[1][r8] > -1e29f) ? __expf(sv[1][r8] - mn) : 0.0f;
        float ps = p0 + p1;
        for (int off = 1; off < 16; off <<= 1) ps += __shfl_xor(ps, off, 16);
        s_r[r8] = s_r[r8] * sc + ps;
        m_r[r8] = mn;
        o0[r8] = o0[r8] * sc;
        o1[r8] = o1[r8] * sc;
        int prow = hi * 8 + r8;
        Pld[wave * 512 + prow * 32 + lcol]      = (__bf16)p0;
        Pld[wave * 512 + prow * 32 + 16 + lcol] = (__bf16)p1;
      }

      // P (A-frag via LDS transpose) x V (B-frags from staged V^T)
      const __bf16* pb  = Pld + wave * 512 + lcol * 32;
      v16bf pf  = frag_bf16(pb + hi * 8, pb + 16 + hi * 8);
      const __bf16* v0p = Vt + (size_t)lcol * 416 + j0 + hi * 8;
      const __bf16* v1p = Vt + (size_t)(lcol + 16) * 416 + j0 + hi * 8;
      v16bf vf0 = frag_bf16(v0p, v0p + 16);
      v16bf vf1 = frag_bf16(v1p, v1p + 16);
      o0 = __builtin_amdgcn_wmma_f32_16x16x32_bf16(false, pf, false, vf0, (short)0, o0, false, false);
      o1 = __builtin_amdgcn_wmma_f32_16x16x32_bf16(false, pf, false, vf1, (short)0, o1, false, false);
    }

#pragma unroll
    for (int r8 = 0; r8 < 8; ++r8) {
      int row = m0 + hi * 8 + r8;
      if (row < 392 && s_r[r8] > 0.0f) {
        float inv = 1.0f / s_r[r8];
        size_t ob = ((size_t)win * 392 + row) * C + head * 32;
        out[ob + lcol]      = (__bf16)(o0[r8] * inv);
        out[ob + 16 + lcol] = (__bf16)(o1[r8] * inv);
      }
    }
  }
}

// ---------------------------------------------------------------------------
// LayerNorm (one wave per token): f32 in, f32 or bf16 out.
// ---------------------------------------------------------------------------
__device__ inline void ln_core(const float* __restrict__ x, float* mo, float* rso,
                               int C, int lane) {
  float s = 0.0f, s2 = 0.0f;
  for (int c = lane; c < C; c += 32) { float v = x[c]; s += v; s2 += v * v; }
  for (int off = 1; off < 32; off <<= 1) { s += __shfl_xor(s, off, 32); s2 += __shfl_xor(s2, off, 32); }
  float m = s / C, var = s2 / C - m * m;
  *mo = m; *rso = rsqrtf(var + 1e-5f);
}

__global__ __launch_bounds__(256)
void ln_f32(const float* __restrict__ in, float* __restrict__ out,
            const float* __restrict__ g, const float* __restrict__ b,
            int tokens, int C) {
  int t = blockIdx.x * 8 + (threadIdx.x >> 5);
  if (t >= tokens) return;
  int lane = threadIdx.x & 31;
  const float* x = in + (size_t)t * C;
  float m, rs; ln_core(x, &m, &rs, C, lane);
  float* o = out + (size_t)t * C;
  for (int c = lane; c < C; c += 32) o[c] = (x[c] - m) * rs * g[c] + b[c];
}

__global__ __launch_bounds__(256)
void ln_b16(const float* __restrict__ in, __bf16* __restrict__ out,
            const float* __restrict__ g, const float* __restrict__ b,
            int tokens, int C) {
  int t = blockIdx.x * 8 + (threadIdx.x >> 5);
  if (t >= tokens) return;
  int lane = threadIdx.x & 31;
  const float* x = in + (size_t)t * C;
  float m, rs; ln_core(x, &m, &rs, C, lane);
  __bf16* o = out + (size_t)t * C;
  for (int c = lane; c < C; c += 32) o[c] = (__bf16)((x[c] - m) * rs * g[c] + b[c]);
}

// ---------------------------------------------------------------------------
// roll(-sh) + window partition gather (bf16) and inverse scatter-add (f32).
// ---------------------------------------------------------------------------
__device__ inline long win_src_index(long wtok, int c, int T, int H, int W, int C,
                                     int s0, int s1, int s2) {
  int win = (int)(wtok / 392), loc = (int)(wtok % 392);
  int nWw = W / 7, nWh = H / 7;
  int ww_ = win % nWw, hw_ = (win / nWw) % nWh, tw_ = win / (nWw * nWh);
  int t = loc / 49, h = (loc % 49) / 7, w = loc % 7;
  int gt = (tw_ * 8 + t + s0) % T;
  int gh = (hw_ * 7 + h + s1) % H;
  int gw = (ww_ * 7 + w + s2) % W;
  return (((long)gt * H + gh) * W + gw) * C + c;
}

__global__ __launch_bounds__(256)
void win_gather_b16(const __bf16* __restrict__ in, __bf16* __restrict__ out,
                    int T, int H, int W, int C, int s0, int s1, int s2) {
  long idx = (long)blockIdx.x * 256 + threadIdx.x;
  long total = (long)T * H * W * C;
  if (idx >= total) return;
  int c = (int)(idx % C);
  out[idx] = in[win_src_index(idx / C, c, T, H, W, C, s0, s1, s2)];
}

__global__ __launch_bounds__(256)
void win_scatter_add(const float* __restrict__ yw, float* __restrict__ x,
                     int T, int H, int W, int C, int s0, int s1, int s2) {
  long idx = (long)blockIdx.x * 256 + threadIdx.x;
  long total = (long)T * H * W * C;
  if (idx >= total) return;
  int c = (int)(idx % C);
  x[win_src_index(idx / C, c, T, H, W, C, s0, s1, s2)] += yw[idx];
}

// ---------------------------------------------------------------------------
// im2col for patch-embed conv (k=stride=(2,4,4)); bf16 out [25088, 96].
// ---------------------------------------------------------------------------
__global__ __launch_bounds__(256)
void im2col_kernel(const float* __restrict__ img, __bf16* __restrict__ out) {
  long idx = (long)blockIdx.x * 256 + threadIdx.x;
  if (idx >= (long)25088 * 96) return;
  int row = (int)(idx / 96), k = (int)(idx % 96);
  int t = row / 3136, rem = row % 3136, h = rem / 56, w = rem % 56;
  int ci = k / 32, r2 = k % 32, dt = r2 / 16, r3 = r2 % 16, dh = r3 / 4, dw = r3 % 4;
  out[idx] = (__bf16)img[(((size_t)ci * 16 + 2 * t + dt) * 224 + 4 * h + dh) * 224 + 4 * w + dw];
}

// ---------------------------------------------------------------------------
// Patch merging concat gather (f32): [x0(0,0), x1(1,0), x2(0,1), x3(1,1)].
// ---------------------------------------------------------------------------
__global__ __launch_bounds__(256)
void merge_gather(const float* __restrict__ x, float* __restrict__ cat,
                  int T, int H, int W, int C) {
  int H2 = H / 2, W2 = W / 2, C4 = 4 * C;
  long total = (long)T * H2 * W2 * C4;
  long idx = (long)blockIdx.x * 256 + threadIdx.x;
  if (idx >= total) return;
  long row = idx / C4; int cp = (int)(idx % C4);
  int which = cp / C, c = cp % C;
  int t = (int)(row / (H2 * W2)), rem = (int)(row % (H2 * W2));
  int h2 = rem / W2, w2 = rem % W2;
  int h = 2 * h2 + (which & 1);
  int w = 2 * w2 + (which >> 1);
  cat[idx] = x[(((long)t * H + h) * W + w) * C + c];
}

// ---------------------------------------------------------------------------
// Cross-attn constant: c2 = pc_w @ vc + pc_b,
// vc = kv_w[C:2C] @ LN_nz(gelu(zw @ LN(text[0]) + zb)) + kv_b[C:2C].
// (identical keys/values => softmax rows sum to 1 => attend(qc,kc,vc) == vc)
// ---------------------------------------------------------------------------
__device__ float block_sum(float v, float* red) {
  for (int off = 1; off < 32; off <<= 1) v += __shfl_xor(v, off, 32);
  int wave = threadIdx.x >> 5;
  if ((threadIdx.x & 31) == 0) red[wave] = v;
  __syncthreads();
  if (threadIdx.x == 0) {
    float t = 0.0f;
    for (int i = 0; i < 8; ++i) t += red[i];
    red[0] = t;
  }
  __syncthreads();
  float out = red[0];
  __syncthreads();
  return out;
}

__global__ __launch_bounds__(256)
void za_kernel(const float* __restrict__ text,
               const float* __restrict__ lng, const float* __restrict__ lnb,
               const float* __restrict__ zw, const float* __restrict__ zb,
               const float* __restrict__ nzg, const float* __restrict__ nzb,
               const float* __restrict__ kvw, const float* __restrict__ kvb,
               const float* __restrict__ pcw, const float* __restrict__ pcb,
               float* __restrict__ c2, int C) {
  __shared__ float zn[512];
  __shared__ float za[768];
  __shared__ float vc[768];
  __shared__ float red[8];
  int tid = threadIdx.x;

  float s = 0.0f, s2 = 0.0f;
  for (int i = tid; i < 512; i += 256) { float v = text[i]; s += v; s2 += v * v; }
  float S = block_sum(s, red), S2 = block_sum(s2, red);
  float mean = S / 512.0f, var = S2 / 512.0f - mean * mean;
  float rs = rsqrtf(var + 1e-5f);
  for (int i = tid; i < 512; i += 256) zn[i] = (text[i] - mean) * rs * lng[i] + lnb[i];
  __syncthreads();

  for (int c = tid; c < C; c += 256) {
    float a = zb[c];
    const float* wr = zw + (size_t)c * 512;
    for (int k = 0; k < 512; ++k) a += wr[k] * zn[k];
    za[c] = gelu_exact(a);
  }
  __syncthreads();

  s = 0.0f; s2 = 0.0f;
  for (int c = tid; c < C; c += 256) { float v = za[c]; s += v; s2 += v * v; }
  S = block_sum(s, red); S2 = block_sum(s2, red);
  mean = S / C; var = S2 / C - mean * mean; rs = rsqrtf(var + 1e-5f);
  for (int c = tid; c < C; c += 256) za[c] = (za[c] - mean) * rs * nzg[c] + nzb[c];
  __syncthreads();

  for (int c = tid; c < C; c += 256) {
    float a = kvb[C + c];
    const float* wr = kvw + (size_t)(C + c) * C;
    for (int k = 0; k < C; ++k) a += wr[k] * za[k];
    vc[c] = a;
  }
  __syncthreads();

  for (int c = tid; c < C; c += 256) {
    float a = pcb[c];
    const float* wr = pcw + (size_t)c * C;
    for (int k = 0; k < C; ++k) a += wr[k] * vc[k];
    c2[c] = a;
  }
}

// ---------------------------------------------------------------------------
// Final mean pool over 392 tokens + classifier head (21 classes).
// ---------------------------------------------------------------------------
__global__ __launch_bounds__(256)
void head_kernel(const float* __restrict__ x, const float* __restrict__ hw,
                 const float* __restrict__ hb, float* __restrict__ out) {
  __shared__ float pooled[768];
  int tid = threadIdx.x;
  for (int c = tid; c < 768; c += 256) {
    float s = 0.0f;
    for (int r = 0; r < 392; ++r) s += x[(size_t)r * 768 + c];
    pooled[c] = s * (1.0f / 392.0f);
  }
  __syncthreads();
  if (tid < 21) {
    float a = hb[tid];
    const float* wr = hw + (size_t)tid * 768;
    for (int k = 0; k < 768; ++k) a += wr[k] * pooled[k];
    out[tid] = a;
  }
}

// ---------------------------------------------------------------------------
// Host orchestration
// ---------------------------------------------------------------------------
extern "C" void kernel_launch(void* const* d_in, const int* in_sizes, int n_in,
                              void* d_out, int out_size, void* d_ws, size_t ws_size,
                              hipStream_t stream) {
  (void)in_sizes; (void)n_in; (void)out_size; (void)ws_size;
  const float* img  = (const float*)d_in[0];
  const float* text = (const float*)d_in[1];
  int pidx = 2;
  auto nxt = [&]() { return (const float*)d_in[pidx++]; };

  const float* pe_w = nxt(); const float* pe_b = nxt();
  const float* pe_ng = nxt(); const float* pe_nb = nxt();

  struct Blk {
    const float *n1_g,*n1_b,*qkv_w,*qkv_b,*proj_w,*proj_b,*rel,*q_w,*q_b,
                *kv_w,*kv_b,*pc_w,*pc_b,*nz_g,*nz_b,*n2_g,*n2_b,
                *fc1_w,*fc1_b,*fc2_w,*fc2_b;
    const __bf16 *qkv_wb,*proj_wb,*fc1_wb,*fc2_wb;
  };
  struct Stage {
    Blk blk[6];
    const float *mg_g,*mg_b,*mg_w;
    const __bf16 *mg_wb;
    const float *pz_g,*pz_b,*pz_w,*pz_bb;
    int depth;
  };
  Stage st[4];
  const int depths[4] = {2, 2, 6, 2};
  const int headsA[4] = {3, 6, 12, 24};
  for (int i = 0; i < 4; ++i) {
    st[i].depth = depths[i];
    for (int j = 0; j < depths[i]; ++j) {
      Blk& b = st[i].blk[j];
      b.n1_g = nxt(); b.n1_b = nxt(); b.qkv_w = nxt(); b.qkv_b = nxt();
      b.proj_w = nxt(); b.proj_b = nxt(); b.rel = nxt();
      b.q_w = nxt(); b.q_b = nxt();            // unused (cross-attn collapses to vc)
      b.kv_w = nxt(); b.kv_b = nxt(); b.pc_w = nxt(); b.pc_b = nxt();
      b.nz_g = nxt(); b.nz_b = nxt(); b.n2_g = nxt(); b.n2_b = nxt();
      b.fc1_w = nxt(); b.fc1_b = nxt(); b.fc2_w = nxt(); b.fc2_b = nxt();
    }
    if (i < 3) { st[i].mg_g = nxt(); st[i].mg_b = nxt(); st[i].mg_w = nxt(); }
    else       { st[i].mg_g = st[i].mg_b = st[i].mg_w = nullptr; }
    st[i].pz_g = nxt(); st[i].pz_b = nxt(); st[i].pz_w = nxt(); st[i].pz_bb = nxt();
  }
  const float* nm_g = nxt(); const float* nm_b = nxt();
  const float* hd_w = nxt(); const float* hd_b = nxt();

  // ---- workspace layout (byte offsets, all 16B aligned) ----
  char* base = (char*)d_ws;
  float*  X    = (float*)(base);                          // 16MB f32
  float*  XNf  = (float*)(base + ((size_t)16 << 20));     // 16MB f32
  float*  Yf   = (float*)(base + ((size_t)32 << 20));     // 16MB f32
  float*  CATf = (float*)(base + ((size_t)48 << 20));     // 16MB f32
  float*  C2   = (float*)(base + ((size_t)64 << 20));     // 4KB f32
  __bf16* XNb  = (__bf16*)(base + ((size_t)65 << 20));    // 8MB bf16
  __bf16* XWb  = (__bf16*)(base + ((size_t)73 << 20));    // 8MB bf16
  __bf16* QKVb = (__bf16*)(base + ((size_t)81 << 20));    // 16MB bf16
  __bf16* ATb  = (__bf16*)(base + ((size_t)97 << 20));    // 8MB bf16
  __bf16* H1b  = (__bf16*)(base + ((size_t)105 << 20));   // 24MB bf16
  __bf16* CATb = (__bf16*)(base + ((size_t)129 << 20));   // 8MB bf16
  __bf16* WB   = (__bf16*)(base + ((size_t)137 << 20));   // 60MB bf16 weight pool

  // ---- one-time weight f32->bf16 pre-pass (L2-resident pool) ----
  __bf16* wb_cursor = WB;
  auto cvtw = [&](const float* w, long n) -> const __bf16* {
    __bf16* dst = wb_cursor;
    cvt_f32_bf16<<<dim3((unsigned)((n + 255) / 256)), dim3(256), 0, stream>>>(w, dst, n);
    wb_cursor += n;
    return dst;
  };
  const __bf16* pe_wb = cvtw(pe_w, (long)96 * 96);
  {
    int Cc = 96;
    for (int i = 0; i < 4; ++i) {
      for (int j = 0; j < st[i].depth; ++j) {
        Blk& b = st[i].blk[j];
        b.qkv_wb  = cvtw(b.qkv_w,  (long)3 * Cc * Cc);
        b.proj_wb = cvtw(b.proj_w, (long)Cc * Cc);
        b.fc1_wb  = cvtw(b.fc1_w,  (long)4 * Cc * Cc);
        b.fc2_wb  = cvtw(b.fc2_w,  (long)4 * Cc * Cc);
      }
      if (i < 3) { st[i].mg_wb = cvtw(st[i].mg_w, (long)2 * Cc * 4 * Cc); Cc *= 2; }
      else         st[i].mg_wb = nullptr;
    }
  }

  auto gemm = [&](const __bf16* A, const __bf16* Wt, const float* bias,
                  const float* res, const float* vec, void* Out,
                  int M, int N, int K, int flags) {
    dim3 g((M + 255) / 256, N / 32);
    gemm_bf16_wmma<<<g, dim3(256), 0, stream>>>(A, Wt, bias, res, vec, Out, M, N, K, flags);
  };
  auto lnf = [&](const float* in, float* out, const float* g_, const float* b_,
                 int tok, int Cn) {
    ln_f32<<<dim3((tok + 7) / 8), dim3(256), 0, stream>>>(in, out, g_, b_, tok, Cn);
  };
  auto lnb = [&](const float* in, __bf16* out, const float* g_, const float* b_,
                 int tok, int Cn) {
    ln_b16<<<dim3((tok + 7) / 8), dim3(256), 0, stream>>>(in, out, g_, b_, tok, Cn);
  };

  // ---- patch embed: im2col -> GEMM -> LN ----
  {
    const int M = 25088;
    im2col_kernel<<<dim3((unsigned)(((long)M * 96 + 255) / 256)), dim3(256), 0, stream>>>(img, QKVb);
    gemm(QKVb, pe_wb, pe_b, nullptr, nullptr, Yf, M, 96, 96, 0);
    lnf(Yf, X, pe_ng, pe_nb, M, 96);
  }

  int T = 8, H = 56, Wd = 56, C = 96;
  for (int i = 0; i < 4; ++i) {
    const int heads = headsA[i];
    const int tok = T * H * Wd;
    const int nW = tok / 392;
    for (int j = 0; j < st[i].depth; ++j) {
      const Blk& b = st[i].blk[j];
      const int s0 = 0;                               // T=8 <= window 8 -> no T shift
      const int s1 = (j % 2 == 1 && H > 7) ? 3 : 0;
      const int s2 = (j % 2 == 1 && Wd > 7) ? 3 : 0;

      za_kernel<<<dim3(1), dim3(256), 0, stream>>>(
          text, st[i].pz_g, st[i].pz_b, st[i].pz_w, st[i].pz_bb,
          b.nz_g, b.nz_b, b.kv_w, b.kv_b, b.pc_w, b.pc_b, C2, C);

      lnb(X, XNb, b.n1_g, b.n1_b, tok, C);
      win_gather_b16<<<dim3((unsigned)(((long)tok * C + 255) / 256)), dim3(256), 0, stream>>>(
          XNb, XWb, T, H, Wd, C, s0, s1, s2);
      gemm(XWb, b.qkv_wb, b.qkv_b, nullptr, nullptr, QKVb, tok, 3 * C, C, 8);
      win_attn<<<dim3(nW * heads), dim3(256), 0, stream>>>(
          QKVb, ATb, b.rel, heads, C, T, H, Wd, s0, s1, s2);
      gemm(ATb, b.proj_wb, b.proj_b, nullptr, C2, Yf, tok, C, C, 4);  // + cross-attn const
      win_scatter_add<<<dim3((unsigned)(((long)tok * C + 255) / 256)), dim3(256), 0, stream>>>(
          Yf, X, T, H, Wd, C, s0, s1, s2);

      lnb(X, XNb, b.n2_g, b.n2_b, tok, C);
      gemm(XNb, b.fc1_wb, b.fc1_b, nullptr, nullptr, H1b, tok, 4 * C, C, 1 | 8);  // gelu, bf16
      gemm(H1b, b.fc2_wb, b.fc2_b, X, nullptr, X, tok, C, 4 * C, 2);              // +residual
    }
    if (i < 3) {
      const int tok2 = T * (H / 2) * (Wd / 2);
      merge_gather<<<dim3((unsigned)(((long)tok2 * 4 * C + 255) / 256)), dim3(256), 0, stream>>>(
          X, CATf, T, H, Wd, C);
      lnb(CATf, CATb, st[i].mg_g, st[i].mg_b, tok2, 4 * C);
      gemm(CATb, st[i].mg_wb, nullptr, nullptr, nullptr, X, tok2, 2 * C, 4 * C, 0);
      H /= 2; Wd /= 2; C *= 2;
    }
  }

  lnf(X, XNf, nm_g, nm_b, 392, 768);
  head_kernel<<<dim3(1), dim3(256), 0, stream>>>(XNf, hd_w, hd_b, (float*)d_out);
}
